// MsgMerger_8126078124687
// MI455X (gfx1250) — matrix operand order repaired and checked by reading, not compiled, not run on hardware
//
#include <hip/hip_runtime.h>
#include <math.h>

#define BB 512
#define NN 256
#define DD 512

typedef __attribute__((ext_vector_type(16))) __bf16        v16bf;
typedef __attribute__((ext_vector_type(2)))  __bf16        v2bf;
typedef __attribute__((ext_vector_type(8)))  float         v8f;
typedef __attribute__((ext_vector_type(8)))  unsigned int  u32x8;
typedef __attribute__((ext_vector_type(4)))  unsigned int  u32x4;

// LDS layout (bytes)
#define OFF_KBUF   0                      // 256*512 bf16  = 262144
#define OFF_QBUF   262144                 // 16*512  bf16  =  16384
#define OFF_SCBUF  278528                 // 16*256  f32   =  16384
#define OFF_RED    294912                 // 256 f32       =   1024
#define OFF_SCOL   295936                 // 256 f32       =   1024
#define OFF_TVEC   296960                 // 512 f32       =   2048
#define OFF_RMAX   299008                 // 16 f32 (pad 64)
#define OFF_RINV   299072                 // 16 f32 (pad 64)
#define SMEM_BYTES 299136

// Workspace layout
#define WS_WQT_OFF   0                                    // 512*512 bf16
#define WS_WKT_OFF   ((size_t)DD * DD)                    // 512*512 bf16
#define WS_MSGB_OFF  ((size_t)2 * DD * DD)                // B*N*D bf16
#define WS_FAST_NEED (((size_t)2 * DD * DD + (size_t)BB * NN * DD) * 2)

__device__ __forceinline__ unsigned pkbf2(float lo, float hi) {
#if __has_builtin(__builtin_amdgcn_cvt_pk_bf16_f32)
    v2bf r = __builtin_amdgcn_cvt_pk_bf16_f32(lo, hi);
    return __builtin_bit_cast(unsigned, r);
#else
    unsigned short a = __builtin_bit_cast(unsigned short, (__bf16)lo);
    unsigned short b = __builtin_bit_cast(unsigned short, (__bf16)hi);
    return (unsigned)a | ((unsigned)b << 16);
#endif
}
__device__ __forceinline__ unsigned short f2bf(float f) {
    return __builtin_bit_cast(unsigned short, (__bf16)f);
}

union FragU {
    unsigned u[8];
    u32x4    h[2];
    u32x8    w;
    v16bf    v;
};

#define WMMA_BF16(A, Bv, C) \
    __builtin_amdgcn_wmma_f32_16x16x32_bf16(false, (A), false, (Bv), (short)0, (C), false, false)

// ---------------- Pre-pass kernels ----------------------------------------

// W[d][e] (f32) -> WT[e][d] (bf16) for Wq and Wk.
__global__ void __launch_bounds__(256)
weights_to_bf16T(const float* __restrict__ Wq, const float* __restrict__ Wk,
                 unsigned short* __restrict__ WqT, unsigned short* __restrict__ WkT) {
    int idx = blockIdx.x * 256 + threadIdx.x;           // 0 .. 512*512-1
    if (idx >= DD * DD) return;
    int e = idx >> 9;
    int d = idx & (DD - 1);
    WqT[(size_t)e * DD + d] = f2bf(Wq[(size_t)d * DD + e]);
    WkT[(size_t)e * DD + d] = f2bf(Wk[(size_t)d * DD + e]);
}

// msg (f32) -> msgb (bf16), 8 elements per thread, b128 in / b128 out.
__global__ void __launch_bounds__(256)
msg_to_bf16(const float* __restrict__ msg, unsigned short* __restrict__ msgb) {
    size_t base = ((size_t)blockIdx.x * 256 + threadIdx.x) * 8;
    const float4* p = (const float4*)(msg + base);
    float4 a = p[0], b = p[1];
    u32x4 o;
    o[0] = pkbf2(a.x, a.y); o[1] = pkbf2(a.z, a.w);
    o[2] = pkbf2(b.x, b.y); o[3] = pkbf2(b.z, b.w);
    *(u32x4*)(msgb + base) = o;
}

// ---------------- A-fragment loaders (ISA 16-bit A 16x32 layout) ----------

template <bool BF16MSG>
__device__ __forceinline__ v16bf load_a_frag(const void* arow, int kc, int lhi) {
    if constexpr (BF16MSG) {
        const unsigned short* p = (const unsigned short*)arow;
        FragU a;
        a.h[0] = *(const u32x4*)(p + kc * 32 + lhi * 8);
        a.h[1] = *(const u32x4*)(p + kc * 32 + lhi * 8 + 16);
        return a.v;
    } else {
        const float* f = (const float*)arow;
        const float4* p0 = (const float4*)(f + kc * 32 + lhi * 8);
        const float4* p1 = (const float4*)(f + kc * 32 + lhi * 8 + 16);
        float4 f0 = p0[0], f1 = p0[1], f2 = p1[0], f3 = p1[1];
        FragU a;
        a.u[0] = pkbf2(f0.x, f0.y); a.u[1] = pkbf2(f0.z, f0.w);
        a.u[2] = pkbf2(f1.x, f1.y); a.u[3] = pkbf2(f1.z, f1.w);
        a.u[4] = pkbf2(f2.x, f2.y); a.u[5] = pkbf2(f2.z, f2.w);
        a.u[6] = pkbf2(f3.x, f3.y); a.u[7] = pkbf2(f3.z, f3.w);
        return a.v;
    }
}

// 16 rows x 64 cols of dst = A(16x512) @ WT^T, 4 e-tiles sharing A-fragments.
template <bool BF16MSG>
__device__ __forceinline__ void
gemm_row_group4(const void* arow, const unsigned short* __restrict__ WT,
                int e0, int lm, int lhi, unsigned short* __restrict__ dst) {
    const unsigned short* br0 = WT + (size_t)(e0      + lm) * DD;
    const unsigned short* br1 = WT + (size_t)(e0 + 16 + lm) * DD;
    const unsigned short* br2 = WT + (size_t)(e0 + 32 + lm) * DD;
    const unsigned short* br3 = WT + (size_t)(e0 + 48 + lm) * DD;
    v8f a0, a1, a2, a3;
    #pragma unroll
    for (int i = 0; i < 8; ++i) { a0[i] = 0.f; a1[i] = 0.f; a2[i] = 0.f; a3[i] = 0.f; }
    #pragma unroll 2
    for (int kc = 0; kc < 16; ++kc) {
        v16bf av = load_a_frag<BF16MSG>(arow, kc, lhi);
        FragU b0, b1, b2, b3;
        b0.h[0] = *(const u32x4*)(br0 + kc * 32 + lhi * 8); b0.h[1] = *(const u32x4*)(br0 + kc * 32 + lhi * 8 + 16);
        b1.h[0] = *(const u32x4*)(br1 + kc * 32 + lhi * 8); b1.h[1] = *(const u32x4*)(br1 + kc * 32 + lhi * 8 + 16);
        b2.h[0] = *(const u32x4*)(br2 + kc * 32 + lhi * 8); b2.h[1] = *(const u32x4*)(br2 + kc * 32 + lhi * 8 + 16);
        b3.h[0] = *(const u32x4*)(br3 + kc * 32 + lhi * 8); b3.h[1] = *(const u32x4*)(br3 + kc * 32 + lhi * 8 + 16);
        a0 = WMMA_BF16(av, b0.v, a0);
        a1 = WMMA_BF16(av, b1.v, a1);
        a2 = WMMA_BF16(av, b2.v, a2);
        a3 = WMMA_BF16(av, b3.v, a3);
    }
    #pragma unroll
    for (int j = 0; j < 8; ++j) {
        const int row = j + lhi * 8;
        dst[row * DD + e0      + lm] = f2bf(a0[j]);
        dst[row * DD + e0 + 16 + lm] = f2bf(a1[j]);
        dst[row * DD + e0 + 32 + lm] = f2bf(a2[j]);
        dst[row * DD + e0 + 48 + lm] = f2bf(a3[j]);
    }
}

// Dual-row variant: 2 independent 16-row A tiles share the 4 B-fragments.
// Per k-step: 2 A-frag loads + 4 B-frag loads -> 8 WMMAs.
template <bool BF16MSG>
__device__ __forceinline__ void
gemm_2row_group4(const void* arow0, const void* arow1,
                 const unsigned short* __restrict__ WT,
                 int e0, int lm, int lhi,
                 unsigned short* __restrict__ dst0,
                 unsigned short* __restrict__ dst1) {
    const unsigned short* br0 = WT + (size_t)(e0      + lm) * DD;
    const unsigned short* br1 = WT + (size_t)(e0 + 16 + lm) * DD;
    const unsigned short* br2 = WT + (size_t)(e0 + 32 + lm) * DD;
    const unsigned short* br3 = WT + (size_t)(e0 + 48 + lm) * DD;
    v8f a0, a1, a2, a3, c0, c1, c2, c3;
    #pragma unroll
    for (int i = 0; i < 8; ++i) {
        a0[i] = 0.f; a1[i] = 0.f; a2[i] = 0.f; a3[i] = 0.f;
        c0[i] = 0.f; c1[i] = 0.f; c2[i] = 0.f; c3[i] = 0.f;
    }
    for (int kc = 0; kc < 16; ++kc) {
        v16bf av0 = load_a_frag<BF16MSG>(arow0, kc, lhi);
        v16bf av1 = load_a_frag<BF16MSG>(arow1, kc, lhi);
        FragU b0, b1, b2, b3;
        b0.h[0] = *(const u32x4*)(br0 + kc * 32 + lhi * 8); b0.h[1] = *(const u32x4*)(br0 + kc * 32 + lhi * 8 + 16);
        b1.h[0] = *(const u32x4*)(br1 + kc * 32 + lhi * 8); b1.h[1] = *(const u32x4*)(br1 + kc * 32 + lhi * 8 + 16);
        b2.h[0] = *(const u32x4*)(br2 + kc * 32 + lhi * 8); b2.h[1] = *(const u32x4*)(br2 + kc * 32 + lhi * 8 + 16);
        b3.h[0] = *(const u32x4*)(br3 + kc * 32 + lhi * 8); b3.h[1] = *(const u32x4*)(br3 + kc * 32 + lhi * 8 + 16);
        a0 = WMMA_BF16(av0, b0.v, a0);
        c0 = WMMA_BF16(av1, b0.v, c0);
        a1 = WMMA_BF16(av0, b1.v, a1);
        c1 = WMMA_BF16(av1, b1.v, c1);
        a2 = WMMA_BF16(av0, b2.v, a2);
        c2 = WMMA_BF16(av1, b2.v, c2);
        a3 = WMMA_BF16(av0, b3.v, a3);
        c3 = WMMA_BF16(av1, b3.v, c3);
    }
    #pragma unroll
    for (int j = 0; j < 8; ++j) {
        const int row = j + lhi * 8;
        dst0[row * DD + e0      + lm] = f2bf(a0[j]);
        dst0[row * DD + e0 + 16 + lm] = f2bf(a1[j]);
        dst0[row * DD + e0 + 32 + lm] = f2bf(a2[j]);
        dst0[row * DD + e0 + 48 + lm] = f2bf(a3[j]);
        dst1[row * DD + e0      + lm] = f2bf(c0[j]);
        dst1[row * DD + e0 + 16 + lm] = f2bf(c1[j]);
        dst1[row * DD + e0 + 32 + lm] = f2bf(c2[j]);
        dst1[row * DD + e0 + 48 + lm] = f2bf(c3[j]);
    }
}

// ---------------- Fused main kernel: one workgroup per batch ---------------
template <bool BF16MSG>
__global__ void __launch_bounds__(256)
msg_merge_kernel(const float* __restrict__ msgf,
                 const unsigned short* __restrict__ msgb,
                 const unsigned short* __restrict__ WqT,
                 const unsigned short* __restrict__ WkT,
                 const float* __restrict__ Wv,
                 float* __restrict__ out) {
    extern __shared__ char smem[];
    unsigned short* kbuf  = (unsigned short*)(smem + OFF_KBUF);
    unsigned short* qbuf  = (unsigned short*)(smem + OFF_QBUF);
    float*          scbuf = (float*)(smem + OFF_SCBUF);
    float*          red   = (float*)(smem + OFF_RED);
    float*          scol  = (float*)(smem + OFF_SCOL);
    float*          tvec  = (float*)(smem + OFF_TVEC);
    float*          rmax  = (float*)(smem + OFF_RMAX);
    float*          rinv  = (float*)(smem + OFF_RINV);

    const int tid  = threadIdx.x;
    const int wid  = tid >> 5;
    const int lane = tid & 31;
    const int lm   = lane & 15;
    const int lhi  = lane >> 4;
    const int b    = blockIdx.x;

    const float*          mbf32 = msgf + (size_t)b * NN * DD;
    const unsigned short* mbf16 = msgb + (size_t)b * NN * DD;

    scol[tid] = 0.0f;

    auto msg_row = [&](int n) -> const void* {
        if constexpr (BF16MSG) return (const void*)(mbf16 + (size_t)n * DD);
        else                   return (const void*)(mbf32 + (size_t)n * DD);
    };

    // -------- Phase A: K = msg[b] @ Wk -> kbuf (bf16 [m][e]) ---------------
    // Each wave owns row-tiles nt=wid and nt=wid+8, processed together so the
    // 4 B-fragments per k-step feed 8 WMMAs.
    {
        const int n0 = wid * 16;
        const int n1 = (wid + 8) * 16;
        const void* arow0 = msg_row(n0 + lm);
        const void* arow1 = msg_row(n1 + lm);
        unsigned short* dst0 = kbuf + (size_t)n0 * DD;
        unsigned short* dst1 = kbuf + (size_t)n1 * DD;
        for (int g = 0; g < 8; ++g)
            gemm_2row_group4<BF16MSG>(arow0, arow1, WkT, g * 64, lm, lhi, dst0, dst1);
    }
    __syncthreads();

    const float norm = 0.044194173824159216f;   // 1/sqrt(512)
    const int r = tid >> 4, c = tid & 15;

    // -------- Phase B: per 16-row q-block ----------------------------------
    for (int blk = 0; blk < 16; ++blk) {
        const int n0 = blk * 16;

        // B1: q_block = msg rows @ Wq -> qbuf; each wave owns 4 e-tiles.
        gemm_row_group4<BF16MSG>(msg_row(n0 + lm), WqT, wid * 64, lm, lhi, qbuf);
        __syncthreads();

        // B2: scores_block(16x256) = q_block @ k^T; each wave owns 2 m-tiles.
        {
            const int m0 = wid * 32;
            const unsigned short* qrow  = qbuf + lm * DD;
            const unsigned short* krow0 = kbuf + (size_t)(m0      + lm) * DD;
            const unsigned short* krow1 = kbuf + (size_t)(m0 + 16 + lm) * DD;
            v8f a0, a1;
            #pragma unroll
            for (int i = 0; i < 8; ++i) { a0[i] = 0.f; a1[i] = 0.f; }
            #pragma unroll 4
            for (int ec = 0; ec < 16; ++ec) {
                FragU af;
                af.h[0] = *(const u32x4*)(qrow + ec * 32 + lhi * 8);
                af.h[1] = *(const u32x4*)(qrow + ec * 32 + lhi * 8 + 16);
                FragU b0, b1;
                b0.h[0] = *(const u32x4*)(krow0 + ec * 32 + lhi * 8); b0.h[1] = *(const u32x4*)(krow0 + ec * 32 + lhi * 8 + 16);
                b1.h[0] = *(const u32x4*)(krow1 + ec * 32 + lhi * 8); b1.h[1] = *(const u32x4*)(krow1 + ec * 32 + lhi * 8 + 16);
                a0 = WMMA_BF16(af.v, b0.v, a0);
                a1 = WMMA_BF16(af.v, b1.v, a1);
            }
            #pragma unroll
            for (int j = 0; j < 8; ++j) {
                scbuf[(j + lhi * 8) * NN + m0      + lm] = a0[j] * norm;
                scbuf[(j + lhi * 8) * NN + m0 + 16 + lm] = a1[j] * norm;
            }
        }
        __syncthreads();

        // Softmax over m (rowwise), accumulate column sums into scol.
        float pm = -3.0e38f;
        for (int m = c; m < NN; m += 16) pm = fmaxf(pm, scbuf[r * NN + m]);
        red[tid] = pm;
        __syncthreads();
        if (tid < 16) {
            float mm = red[tid * 16];
            #pragma unroll
            for (int i = 1; i < 16; ++i) mm = fmaxf(mm, red[tid * 16 + i]);
            rmax[tid] = mm;
        }
        __syncthreads();
        const float rm = rmax[r];
        float ps = 0.0f;
        for (int m = c; m < NN; m += 16) {
            float e = __expf(scbuf[r * NN + m] - rm);
            scbuf[r * NN + m] = e;
            ps += e;
        }
        red[tid] = ps;
        __syncthreads();
        if (tid < 16) {
            float ss = 0.0f;
            #pragma unroll
            for (int i = 0; i < 16; ++i) ss += red[tid * 16 + i];
            rinv[tid] = 1.0f / ss;
        }
        __syncthreads();
        float accs = 0.0f;
        #pragma unroll
        for (int rr = 0; rr < 16; ++rr) accs += scbuf[rr * NN + tid] * rinv[rr];
        scol[tid] += accs;
        __syncthreads();
    }

    // -------- Phase C: t[d] = sum_m scol[m] * msg[b][m][d] -----------------
    float t0 = 0.0f, t1 = 0.0f;
    for (int m = 0; m < NN; ++m) {
        const float sm = scol[m];
        if constexpr (BF16MSG) {
            const unsigned short* mr = mbf16 + (size_t)m * DD;
            t0 += sm * (float)__builtin_bit_cast(__bf16, mr[tid]);
            t1 += sm * (float)__builtin_bit_cast(__bf16, mr[tid + 256]);
        } else {
            const float* mr = mbf32 + (size_t)m * DD;
            t0 += sm * mr[tid];
            t1 += sm * mr[tid + 256];
        }
    }
    tvec[tid] = t0;
    tvec[tid + 256] = t1;
    __syncthreads();

    // -------- Phase D: merged[b][:] = t @ Wv -------------------------------
    float a0 = 0.0f, a1 = 0.0f;
    for (int d = 0; d < DD; ++d) {
        const float tv = tvec[d];
        a0 += tv * Wv[(size_t)d * DD + tid];
        a1 += tv * Wv[(size_t)d * DD + tid + 256];
    }
    out[(size_t)b * DD + tid]       = a0;
    out[(size_t)b * DD + tid + 256] = a1;
}

extern "C" void kernel_launch(void* const* d_in, const int* in_sizes, int n_in,
                              void* d_out, int out_size, void* d_ws, size_t ws_size,
                              hipStream_t stream) {
    const float* msg = (const float*)d_in[0];
    const float* Wq  = (const float*)d_in[1];
    const float* Wk  = (const float*)d_in[2];
    const float* Wv  = (const float*)d_in[3];

    unsigned short* wsb  = (unsigned short*)d_ws;
    unsigned short* WqT  = wsb + WS_WQT_OFF;
    unsigned short* WkT  = wsb + WS_WKT_OFF;
    unsigned short* msgb = wsb + WS_MSGB_OFF;

    weights_to_bf16T<<<(DD * DD + 255) / 256, 256, 0, stream>>>(Wq, Wk, WqT, WkT);

    if (ws_size >= WS_FAST_NEED) {
        (void)hipFuncSetAttribute((const void*)&msg_merge_kernel<true>,
                                  hipFuncAttributeMaxDynamicSharedMemorySize,
                                  SMEM_BYTES);
        const size_t nelem = (size_t)BB * NN * DD;
        msg_to_bf16<<<(unsigned)(nelem / (256 * 8)), 256, 0, stream>>>(msg, msgb);
        msg_merge_kernel<true><<<BB, 256, SMEM_BYTES, stream>>>(
            msg, msgb, WqT, WkT, Wv, (float*)d_out);
    } else {
        (void)hipFuncSetAttribute((const void*)&msg_merge_kernel<false>,
                                  hipFuncAttributeMaxDynamicSharedMemorySize,
                                  SMEM_BYTES);
        msg_merge_kernel<false><<<BB, 256, SMEM_BYTES, stream>>>(
            msg, (const unsigned short*)msg /*unused*/, WqT, WkT, Wv, (float*)d_out);
    }
}
